// OuterEncDec_33071248180097
// MI455X (gfx1250) — matrix operand order repaired and checked by reading, not compiled, not run on hardware
//
#include <hip/hip_runtime.h>
#include <math.h>

typedef _Float16 f16;
typedef _Float16 v16h __attribute__((ext_vector_type(16)));
typedef _Float16 v8h  __attribute__((ext_vector_type(8)));
typedef float    v8f  __attribute__((ext_vector_type(8)));

#define CAT16(lo, hi)                                                          \
  __builtin_shufflevector(lo, hi, 0, 1, 2, 3, 4, 5, 6, 7, 8, 9, 10, 11, 12,   \
                          13, 14, 15)

// ------------------------------------------------------------------
// WMMA: D = A(16x32 f16) * B(32x16 f16) + C(f32)
// A frag: lane l<16 row l, K {0..7,16..23}; lane l+16 same row, K {8..15,24..31}
// B frag (prepacked): lane l<16 col l (K {0..7,16..23}), lane l+16 col l-16 (K {8..15,24..31}),
//                     element e of the lane = row K0+8h+(e<8 ? e : e+8), h = l/16  -> contiguous 32B/lane
// C frag: lane l<16 col l rows 0..7; lane l+16 col l-16 rows 8..15
// ------------------------------------------------------------------
__device__ __forceinline__ v8f wmma_f16(v16h a, v16h b, v8f c) {
  return __builtin_amdgcn_wmma_f32_16x16x32_f16(false, a, false, b, (short)0, c,
                                                false, false);
}

// ---------------- generic WMMA GEMM over prepacked B -------------------------
// A: [M,Kp] f16 row-major (M % 16 == 0). Bp: packed [Np/16][Kp/32][512].
// Each wave: 4 M-tiles x 1 N-tile, ALL wmmas unconditional (rows clamped,
// stores guarded) so accumulators never need phis/copies.
__global__ void __launch_bounds__(32)
gemm_f16_wmma(const f16* __restrict__ A, int M, int Kp,
              const f16* __restrict__ Bp, int Np,
              const float* __restrict__ bias, const float* __restrict__ alpha,
              f16* __restrict__ outh, float* __restrict__ outf, int ldf,
              int ncols, int epilog) {
  const int lane = threadIdx.x;
  const int l15 = lane & 15;
  const int hb = (lane >> 4) & 1;
  const int tg = blockIdx.x, tn = blockIdx.y;
  const int col = tn * 16 + l15;
  const int kChunks = Kp >> 5;
  const int Mtiles = M >> 4;
  // clamped row base for each subtile (always a valid address)
  int arow[4];
#pragma unroll
  for (int s = 0; s < 4; ++s) {
    int tm = tg * 4 + s;
    if (tm > Mtiles - 1) tm = Mtiles - 1;
    arow[s] = tm * 16 + l15;
  }
  v8f c0 = {}, c1 = {}, c2 = {}, c3 = {};
  for (int kci = 0; kci < kChunks; ++kci) {
    const f16* bp = Bp + ((((size_t)tn * kChunks + kci) << 9)) + lane * 16;
    v8h blo = *(const v8h*)bp;
    v8h bhi = *(const v8h*)(bp + 8);
    v16h b = CAT16(blo, bhi);
    const int ko = (kci << 5) + hb * 8;
    {
      const f16* ap = A + (size_t)arow[0] * Kp + ko;
      v16h a = CAT16(*(const v8h*)ap, *(const v8h*)(ap + 16));
      c0 = wmma_f16(a, b, c0);
    }
    {
      const f16* ap = A + (size_t)arow[1] * Kp + ko;
      v16h a = CAT16(*(const v8h*)ap, *(const v8h*)(ap + 16));
      c1 = wmma_f16(a, b, c1);
    }
    {
      const f16* ap = A + (size_t)arow[2] * Kp + ko;
      v16h a = CAT16(*(const v8h*)ap, *(const v8h*)(ap + 16));
      c2 = wmma_f16(a, b, c2);
    }
    {
      const f16* ap = A + (size_t)arow[3] * Kp + ko;
      v16h a = CAT16(*(const v8h*)ap, *(const v8h*)(ap + 16));
      c3 = wmma_f16(a, b, c3);
    }
  }
  const float al = alpha ? alpha[0] : 0.f;
  const float bs = bias ? bias[col] : 0.f;
  v8f acc[4] = {c0, c1, c2, c3};
#pragma unroll
  for (int s = 0; s < 4; ++s) {
    int tm = tg * 4 + s;
    if (tm >= Mtiles) break;   // store-only guard (uniform)
#pragma unroll
    for (int r = 0; r < 8; ++r) {
      int m = tm * 16 + hb * 8 + r;
      float v = acc[s][r] + bs;
      if (epilog == 0) v = (v >= 0.f) ? v : al * v;
      else if (epilog == 1) v = tanhf(v);
      if (outh) outh[(size_t)m * Np + col] = (f16)v;
      if (outf && col < ncols) outf[(size_t)m * ldf + col] = v;
    }
  }
}

// ---------------- implicit-GEMM conv3x3 (stride 1, pad 1), HWC f16 -----------
// act: [H*W, Cinp] f16.  wgt packed: [9][Coutp/16][Cinp/32][512] (flip folded).
// Each wave: 64 pixels (4 M-tiles) x 16 out-channels; all wmmas unconditional,
// boundary handled by per-lane zeroed A loads, stores guarded.
__global__ void __launch_bounds__(32)
conv3_wmma(const f16* __restrict__ act, int H, int W, int Cinp,
           const f16* __restrict__ wgt, int Coutp,
           const float* __restrict__ bias, const float* __restrict__ alpha,
           f16* __restrict__ outh, float* __restrict__ outf, int epilog) {
  const int lane = threadIdx.x;
  const int l15 = lane & 15;
  const int hb = (lane >> 4) & 1;
  const int HW = H * W;
  const int tn = blockIdx.y;
  const int col = tn * 16 + l15;
  const int nTiles = Coutp >> 4;
  const int kChunks = Cinp >> 5;
  int py[4], px[4];
  bool pok[4];
#pragma unroll
  for (int s = 0; s < 4; ++s) {
    int p = (blockIdx.x * 4 + s) * 16 + l15;
    pok[s] = (p < HW);
    int pc = pok[s] ? p : 0;
    py[s] = pc / W;
    px[s] = pc % W;
  }
  v8f c0 = {}, c1 = {}, c2 = {}, c3 = {};
  for (int tap = 0; tap < 9; ++tap) {
    const int ky = tap / 3, kx = tap % 3;
    for (int kci = 0; kci < kChunks; ++kci) {
      const f16* bp =
          wgt + ((((size_t)tap * nTiles + tn) * kChunks + kci) << 9) + lane * 16;
      v8h blo = *(const v8h*)bp;
      v8h bhi = *(const v8h*)(bp + 8);
      v16h b = CAT16(blo, bhi);
      const int ko = (kci << 5) + hb * 8;
#define CONV_SUB(S, CC)                                                        \
      {                                                                        \
        int yy = py[S] + ky - 1, xx = px[S] + kx - 1;                          \
        bool valid = pok[S] && yy >= 0 && yy < H && xx >= 0 && xx < W;         \
        v16h a = {};                                                           \
        if (valid) {                                                           \
          const f16* ap = act + ((size_t)yy * W + xx) * Cinp + ko;             \
          a = CAT16(*(const v8h*)ap, *(const v8h*)(ap + 16));                  \
        }                                                                      \
        CC = wmma_f16(a, b, CC);                                               \
      }
      CONV_SUB(0, c0)
      CONV_SUB(1, c1)
      CONV_SUB(2, c2)
      CONV_SUB(3, c3)
#undef CONV_SUB
    }
  }
  const float al = alpha ? alpha[0] : 0.f;
  const float bs = bias ? bias[col] : 0.f;
  v8f acc[4] = {c0, c1, c2, c3};
#pragma unroll
  for (int s = 0; s < 4; ++s) {
#pragma unroll
    for (int r = 0; r < 8; ++r) {
      int m = (blockIdx.x * 4 + s) * 16 + hb * 8 + r;
      if (m >= HW) continue;
      float v = acc[s][r] + bs;
      if (epilog == 0) v = (v >= 0.f) ? v : al * v;
      else if (epilog == 1) v = tanhf(v);
      if (outh) outh[(size_t)m * Coutp + col] = (f16)v;
      if (outf) outf[(size_t)m * Coutp + col] = v;
    }
  }
}

// ---------------- pooling / upsample / utility kernels -----------------------
__global__ void maxpool2_k(const f16* __restrict__ in, f16* __restrict__ out,
                           int Ho, int Wo, int C) {
  int i = blockIdx.x * blockDim.x + threadIdx.x;
  int total = Ho * Wo * C;
  if (i >= total) return;
  int c = i % C; int pp = i / C; int x = pp % Wo; int y = pp / Wo;
  int Wi = 2 * Wo;
  size_t b = ((size_t)(2 * y) * Wi + 2 * x) * C + c;
  float v0 = (float)in[b], v1 = (float)in[b + C];
  float v2 = (float)in[b + (size_t)Wi * C], v3 = (float)in[b + (size_t)Wi * C + C];
  out[i] = (f16)fmaxf(fmaxf(v0, v1), fmaxf(v2, v3));
}

__global__ void up2_k(const f16* __restrict__ in, f16* __restrict__ out,
                      int Hi, int Wi, int C) {
  int i = blockIdx.x * blockDim.x + threadIdx.x;
  int Wo = 2 * Wi;
  int total = 2 * Hi * Wo * C;
  if (i >= total) return;
  int c = i % C; int pp = i / C; int x = pp % Wo; int y = pp / Wo;
  out[i] = in[((size_t)(y >> 1) * Wi + (x >> 1)) * C + c];
}

__global__ void zero_f32_k(float* p, int n) {
  int i = blockIdx.x * blockDim.x + threadIdx.x;
  if (i < n) p[i] = 0.f;
}
__global__ void zero_f16_k(f16* p, int n) {
  int i = blockIdx.x * blockDim.x + threadIdx.x;
  if (i < n) p[i] = (f16)0.f;
}

// pad+convert activations (row major [M,K] f32 -> [M,Kp] f16)
__global__ void padcvt_x_k(const float* __restrict__ x, f16* __restrict__ xh,
                           int M, int K, int Kp) {
  int i = blockIdx.x * blockDim.x + threadIdx.x;
  if (i >= M * Kp) return;
  int r = i / Kp, k = i % Kp;
  xh[i] = (k < K) ? (f16)x[(size_t)r * K + k] : (f16)0.f;
}

// MLP weight raw [K,N] f32 -> packed B fragments [Np/16][Kp/32][512] f16
__global__ void prep_w_pack_k(const float* __restrict__ raw,
                              f16* __restrict__ dst, int K, int N, int Kp,
                              int Np) {
  int i = blockIdx.x * blockDim.x + threadIdx.x;
  if (i >= Kp * Np) return;
  int e = i & 15;
  int lane = (i >> 4) & 31;
  int chunk = i >> 9;
  int kChunks = Kp >> 5;
  int ntile = chunk / kChunks;
  int kci = chunk % kChunks;
  int col = ntile * 16 + (lane & 15);
  int krow = (kci << 5) + ((lane & 16) ? 8 : 0) + (e < 8 ? e : e + 8);
  dst[i] = (krow < K && col < N) ? (f16)raw[(size_t)krow * N + col] : (f16)0.f;
}

__global__ void prep_b_k(const float* __restrict__ raw, float* __restrict__ dst,
                         int N, int Np) {
  int n = blockIdx.x * blockDim.x + threadIdx.x;
  if (n < Np) dst[n] = (n < N) ? raw[n] : 0.f;
}

// conv weight -> packed [9][Coutp/16][Cinp/32][512] f16
// enc raw OIHW (Cout,Cin,3,3); dec raw (Cin,Cout,3,3) with 180-degree flip.
__global__ void prep_cw_pack_k(const float* __restrict__ raw,
                               f16* __restrict__ dst, int Cin, int Cout,
                               int Cinp, int Coutp, int transposed) {
  int i = blockIdx.x * blockDim.x + threadIdx.x;
  int per_tap = Cinp * Coutp;
  if (i >= 9 * per_tap) return;
  int tap = i / per_tap;
  int r = i % per_tap;
  int e = r & 15;
  int lane = (r >> 4) & 31;
  int chunk = r >> 9;
  int kChunks = Cinp >> 5;
  int ntile = chunk / kChunks;
  int kci = chunk % kChunks;
  int co = ntile * 16 + (lane & 15);
  int ci = (kci << 5) + ((lane & 16) ? 8 : 0) + (e < 8 ? e : e + 8);
  int ky = tap / 3, kx = tap % 3;
  float v = 0.f;
  if (ci < Cin && co < Cout) {
    if (transposed)
      v = raw[(((size_t)ci * Cout + co) * 3 + (2 - ky)) * 3 + (2 - kx)];
    else
      v = raw[(((size_t)co * Cin + ci) * 3 + ky) * 3 + kx];
  }
  dst[i] = (f16)v;
}

__global__ void enc_norm_k(const float* __restrict__ enc_raw,
                           float* __restrict__ out_enc,
                           float* __restrict__ enc_n, int M) {
  int r = blockIdx.x * blockDim.x + threadIdx.x;
  if (r >= M) return;
  float s = 0.f;
  for (int e = 0; e < 10; ++e) { float v = enc_raw[r * 16 + e]; s += v * v; }
  float inv = 1.f / (sqrtf(s) + 1e-10f);
  for (int e = 0; e < 10; ++e) {
    float v = enc_raw[r * 16 + e] * inv;
    out_enc[r * 10 + e] = v;
    enc_n[r * 10 + e] = v;
  }
}

__global__ void din_norm_k(const float* __restrict__ din_raw,
                           float* __restrict__ out_din,
                           f16* __restrict__ din_h, int M) {
  int r = blockIdx.x * blockDim.x + threadIdx.x;
  if (r >= M) return;
  float s = 0.f;
  for (int e = 0; e < 10; ++e) { float v = din_raw[r * 10 + e]; s += v * v; }
  float inv = 1.f / (sqrtf(s) + 1e-10f);
  for (int e = 0; e < 32; ++e) {
    if (e < 10) {
      float v = din_raw[r * 10 + e] * inv;
      out_din[r * 10 + e] = v;
      din_h[r * 32 + e] = (f16)v;
    } else {
      din_h[r * 32 + e] = (f16)0.f;
    }
  }
}

__global__ void scatter_base_k(const int* __restrict__ inx_s,
                               const int* __restrict__ y,
                               const float* __restrict__ enc_n,
                               float* __restrict__ img_out,
                               f16* __restrict__ a0, int n) {
  int j = blockIdx.x * blockDim.x + threadIdx.x;
  if (j >= n) return;
  int idx = inx_s[j];
  int ccol = y[idx * 2 + 0], rrow = y[idx * 2 + 1];
  size_t pp = (size_t)rrow * 360 + ccol;
  for (int e = 0; e < 10; ++e) {
    float v = enc_n[idx * 10 + e];
    img_out[(size_t)e * 129600 + pp] = v;
    a0[pp * 32 + e] = (f16)v;
  }
  img_out[(size_t)10 * 129600 + pp] = 1.f;
  a0[pp * 32 + 10] = (f16)1.f;
}

__global__ void poke_k(const int* __restrict__ inx_v2, const int* __restrict__ y,
                       const float* __restrict__ enc_n, f16* __restrict__ a0,
                       f16* __restrict__ save, int i) {
  int e = threadIdx.x;
  if (e >= 11) return;
  int idx = inx_v2[i];
  int ccol = y[idx * 2 + 0], rrow = y[idx * 2 + 1];
  size_t off = ((size_t)rrow * 360 + ccol) * 32 + e;
  save[e] = a0[off];
  a0[off] = (e < 10) ? (f16)enc_n[idx * 10 + e] : (f16)1.f;
}

__global__ void restore_k(const int* __restrict__ inx_v2,
                          const int* __restrict__ y, f16* __restrict__ a0,
                          const f16* __restrict__ save, int i) {
  int e = threadIdx.x;
  if (e >= 11) return;
  int idx = inx_v2[i];
  int ccol = y[idx * 2 + 0], rrow = y[idx * 2 + 1];
  a0[((size_t)rrow * 360 + ccol) * 32 + e] = save[e];
}

__global__ void gather_d1_k(const int* __restrict__ inx_v1,
                            const int* __restrict__ y,
                            const float* __restrict__ dout3,
                            float* __restrict__ din_raw, int n) {
  int j = blockIdx.x * blockDim.x + threadIdx.x;
  if (j >= n) return;
  int idx = inx_v1[j];
  int ccol = y[idx * 2 + 0], rrow = y[idx * 2 + 1];
  const float* src = dout3 + ((size_t)rrow * 360 + ccol) * 16;
  for (int e = 0; e < 10; ++e) din_raw[j * 10 + e] = src[e];
}

__global__ void gather_d2_k(const int* __restrict__ inx_v2,
                            const int* __restrict__ y,
                            const float* __restrict__ dout3,
                            float* __restrict__ din_raw, int i) {
  int e = threadIdx.x;
  if (e >= 10) return;
  int idx = inx_v2[i];
  int ccol = y[idx * 2 + 0], rrow = y[idx * 2 + 1];
  din_raw[(512 + i) * 10 + e] = dout3[((size_t)rrow * 360 + ccol) * 16 + e];
}

// ------------------------------------------------------------------
extern "C" void kernel_launch(void* const* d_in, const int* in_sizes, int n_in,
                              void* d_out, int out_size, void* d_ws,
                              size_t ws_size, hipStream_t stream) {
  (void)in_sizes; (void)n_in; (void)out_size; (void)ws_size;

  // ---- input pointers (params in dict insertion order, then x,y,indices) ----
  const float* enc_w[4] = {(const float*)d_in[0], (const float*)d_in[3],
                           (const float*)d_in[6], (const float*)d_in[9]};
  const float* enc_b[4] = {(const float*)d_in[1], (const float*)d_in[4],
                           (const float*)d_in[7], (const float*)d_in[10]};
  const float* enc_a[3] = {(const float*)d_in[2], (const float*)d_in[5],
                           (const float*)d_in[8]};
  const float* dec_w[4] = {(const float*)d_in[11], (const float*)d_in[14],
                           (const float*)d_in[17], (const float*)d_in[20]};
  const float* dec_b[4] = {(const float*)d_in[12], (const float*)d_in[15],
                           (const float*)d_in[18], (const float*)d_in[21]};
  const float* dec_a[3] = {(const float*)d_in[13], (const float*)d_in[16],
                           (const float*)d_in[19]};
  const float* ae_w[8] = {(const float*)d_in[22], (const float*)d_in[25],
                          (const float*)d_in[28], (const float*)d_in[31],
                          (const float*)d_in[34], (const float*)d_in[37],
                          (const float*)d_in[40], (const float*)d_in[43]};
  const float* ae_b[8] = {(const float*)d_in[23], (const float*)d_in[26],
                          (const float*)d_in[29], (const float*)d_in[32],
                          (const float*)d_in[35], (const float*)d_in[38],
                          (const float*)d_in[41], (const float*)d_in[44]};
  const float* ae_a[7] = {(const float*)d_in[24], (const float*)d_in[27],
                          (const float*)d_in[30], (const float*)d_in[33],
                          (const float*)d_in[36], (const float*)d_in[39],
                          (const float*)d_in[42]};
  const float* x = (const float*)d_in[45];
  const int* y = (const int*)d_in[46];
  const int* inx_s = (const int*)d_in[47];
  const int* inx_v1 = (const int*)d_in[48];
  const int* inx_v2 = (const int*)d_in[49];

  float* out = (float*)d_out;
  float* out_enc = out;                 // 4096 x 10
  float* out_din = out + 40960;         // 528 x 10
  float* out_dec = out + 46240;         // 528 x 57
  float* out_img = out + 76336;         // 11 x 360 x 360
  const int OUT_TOTAL = 1501936;

  // ---- workspace bump allocator ----
  char* wsb = (char*)d_ws;
  size_t off = 0;
  auto alloc = [&](size_t bytes) -> void* {
    off = (off + 255) & ~(size_t)255;
    void* p = wsb + off;
    off += bytes;
    return p;
  };

  const int HW = 360 * 360;
  // MLP dims
  const int eK[4] = {285, 256, 128, 64}, eN[4] = {256, 128, 64, 10};
  const int eKp[4] = {288, 256, 128, 64}, eNp[4] = {256, 128, 64, 16};
  const int dK[4] = {10, 64, 128, 256}, dN[4] = {64, 128, 256, 57};
  const int dKp[4] = {32, 64, 128, 256}, dNp[4] = {64, 128, 256, 64};
  // AE dims (e0..e3, d0..d3)
  const int cCin[8] = {11, 32, 64, 128, 256, 128, 64, 32};
  const int cCout[8] = {32, 64, 128, 256, 128, 64, 32, 10};
  const int cCinp[8] = {32, 32, 64, 128, 256, 128, 64, 32};
  const int cCoutp[8] = {32, 64, 128, 256, 128, 64, 32, 16};

  f16* xh = (f16*)alloc((size_t)4096 * 288 * 2);
  f16* ew[4]; float* ebp[4];
  for (int i = 0; i < 4; ++i) {
    ew[i] = (f16*)alloc((size_t)eKp[i] * eNp[i] * 2);
    ebp[i] = (float*)alloc((size_t)eNp[i] * 4);
  }
  f16* dw[4]; float* dbp[4];
  for (int i = 0; i < 4; ++i) {
    dw[i] = (f16*)alloc((size_t)dKp[i] * dNp[i] * 2);
    dbp[i] = (float*)alloc((size_t)dNp[i] * 4);
  }
  f16* cw[8]; float* cbp[8];
  for (int i = 0; i < 8; ++i) {
    cw[i] = (f16*)alloc((size_t)9 * cCinp[i] * cCoutp[i] * 2);
    cbp[i] = (float*)alloc((size_t)cCoutp[i] * 4);
  }
  f16* h0 = (f16*)alloc((size_t)4096 * 256 * 2);
  f16* h1 = (f16*)alloc((size_t)4096 * 128 * 2);
  f16* h2 = (f16*)alloc((size_t)4096 * 64 * 2);
  float* enc_raw = (float*)alloc((size_t)4096 * 16 * 4);
  float* enc_n = (float*)alloc((size_t)4096 * 10 * 4);
  f16* a0 = (f16*)alloc((size_t)HW * 32 * 2);
  f16* P = (f16*)alloc((size_t)HW * 32 * 2);
  f16* Q = (f16*)alloc((size_t)180 * 180 * 32 * 2);
  float* dout3 = (float*)alloc((size_t)HW * 16 * 4);
  float* din_raw = (float*)alloc((size_t)528 * 10 * 4);
  f16* din_h = (f16*)alloc((size_t)528 * 32 * 2);
  f16* g0 = (f16*)alloc((size_t)528 * 64 * 2);
  f16* g1 = (f16*)alloc((size_t)528 * 128 * 2);
  f16* g2 = (f16*)alloc((size_t)528 * 256 * 2);
  f16* save = (f16*)alloc(32 * 2);

  auto eg = [](int n) { return dim3((unsigned)((n + 255) / 256)); };

  // ---- zero output + base image channels ----
  zero_f32_k<<<eg(OUT_TOTAL), 256, 0, stream>>>(out, OUT_TOTAL);
  zero_f16_k<<<eg(HW * 32), 256, 0, stream>>>(a0, HW * 32);

  // ---- weight/bias prep (pack B fragments for WMMA) ----
  padcvt_x_k<<<eg(4096 * 288), 256, 0, stream>>>(x, xh, 4096, 285, 288);
  for (int i = 0; i < 4; ++i) {
    prep_w_pack_k<<<eg(eKp[i] * eNp[i]), 256, 0, stream>>>(
        enc_w[i], ew[i], eK[i], eN[i], eKp[i], eNp[i]);
    prep_b_k<<<eg(eNp[i]), 256, 0, stream>>>(enc_b[i], ebp[i], eN[i], eNp[i]);
    prep_w_pack_k<<<eg(dKp[i] * dNp[i]), 256, 0, stream>>>(
        dec_w[i], dw[i], dK[i], dN[i], dKp[i], dNp[i]);
    prep_b_k<<<eg(dNp[i]), 256, 0, stream>>>(dec_b[i], dbp[i], dN[i], dNp[i]);
  }
  for (int i = 0; i < 8; ++i) {
    prep_cw_pack_k<<<eg(9 * cCinp[i] * cCoutp[i]), 256, 0, stream>>>(
        ae_w[i], cw[i], cCin[i], cCout[i], cCinp[i], cCoutp[i], (i >= 4) ? 1 : 0);
    prep_b_k<<<eg(cCoutp[i]), 256, 0, stream>>>(ae_b[i], cbp[i], cCout[i],
                                                cCoutp[i]);
  }

  auto gemm = [&](const f16* A, int M, int Kp, const f16* B, int Np,
                  const float* bias, const float* alpha, f16* oh, float* of,
                  int ldf, int ncols, int epi) {
    int mgroups = (M / 16 + 3) / 4;
    dim3 g((unsigned)mgroups, (unsigned)(Np / 16));
    gemm_f16_wmma<<<g, 32, 0, stream>>>(A, M, Kp, B, Np, bias, alpha, oh, of,
                                        ldf, ncols, epi);
  };

  // ---- enc MLP: 4096 x (285->256->128->64->10) ----
  gemm(xh, 4096, 288, ew[0], 256, ebp[0], enc_a[0], h0, nullptr, 0, 0, 0);
  gemm(h0, 4096, 256, ew[1], 128, ebp[1], enc_a[1], h1, nullptr, 0, 0, 0);
  gemm(h1, 4096, 128, ew[2], 64, ebp[2], enc_a[2], h2, nullptr, 0, 0, 0);
  gemm(h2, 4096, 64, ew[3], 16, ebp[3], nullptr, nullptr, enc_raw, 16, 16, 1);
  enc_norm_k<<<eg(4096), 256, 0, stream>>>(enc_raw, out_enc, enc_n, 4096);

  // ---- build base image (f32 CHW output + f16 HWC activation) ----
  scatter_base_k<<<eg(2048), 256, 0, stream>>>(inx_s, y, enc_n, out_img, a0,
                                               2048);

  auto conv = [&](const f16* in, int H, int W, int ci, const f16* wp,
                  const float* bp, const float* ap, f16* oh, float* of, int co,
                  int epi) {
    int groups = ((H * W + 15) / 16 + 3) / 4;
    dim3 g((unsigned)groups, (unsigned)(co / 16));
    conv3_wmma<<<g, 32, 0, stream>>>(in, H, W, ci, wp, co, bp, ap, oh, of, epi);
  };
  auto pool = [&](const f16* in, f16* o, int Ho, int Wo, int C) {
    maxpool2_k<<<eg(Ho * Wo * C), 256, 0, stream>>>(in, o, Ho, Wo, C);
  };
  auto up2 = [&](const f16* in, f16* o, int Hi, int Wi, int C) {
    up2_k<<<eg(4 * Hi * Wi * C), 256, 0, stream>>>(in, o, Hi, Wi, C);
  };

  auto ae_forward = [&]() {
    conv(a0, 360, 360, 32, cw[0], cbp[0], ae_a[0], P, nullptr, 32, 0);
    pool(P, Q, 180, 180, 32);
    conv(Q, 180, 180, 32, cw[1], cbp[1], ae_a[1], P, nullptr, 64, 0);
    pool(P, Q, 90, 90, 64);
    conv(Q, 90, 90, 64, cw[2], cbp[2], ae_a[2], P, nullptr, 128, 0);
    pool(P, Q, 45, 45, 128);
    conv(Q, 45, 45, 128, cw[3], cbp[3], ae_a[3], P, nullptr, 256, 0);
    conv(P, 45, 45, 256, cw[4], cbp[4], ae_a[4], Q, nullptr, 128, 0);
    up2(Q, P, 45, 45, 128);
    conv(P, 90, 90, 128, cw[5], cbp[5], ae_a[5], Q, nullptr, 64, 0);
    up2(Q, P, 90, 90, 64);
    conv(P, 180, 180, 64, cw[6], cbp[6], ae_a[6], Q, nullptr, 32, 0);
    up2(Q, P, 180, 180, 32);
    conv(P, 360, 360, 32, cw[7], cbp[7], nullptr, nullptr, dout3, 16, 1);
  };

  // ---- pass 1: interp = AE(img); gather d1 at 512 points ----
  ae_forward();
  gather_d1_k<<<eg(512), 256, 0, stream>>>(inx_v1, y, dout3, din_raw, 512);

  // ---- 16 single-pixel variants: poke -> AE -> gather -> restore ----
  for (int i = 0; i < 16; ++i) {
    poke_k<<<1, 32, 0, stream>>>(inx_v2, y, enc_n, a0, save, i);
    ae_forward();
    gather_d2_k<<<1, 32, 0, stream>>>(inx_v2, y, dout3, din_raw, i);
    restore_k<<<1, 32, 0, stream>>>(inx_v2, y, a0, save, i);
  }

  // ---- din normalize + dec MLP: 528 x (10->64->128->256->57) ----
  din_norm_k<<<eg(528), 256, 0, stream>>>(din_raw, out_din, din_h, 528);
  gemm(din_h, 528, 32, dw[0], 64, dbp[0], dec_a[0], g0, nullptr, 0, 0, 0);
  gemm(g0, 528, 64, dw[1], 128, dbp[1], dec_a[1], g1, nullptr, 0, 0, 0);
  gemm(g1, 528, 128, dw[2], 256, dbp[2], dec_a[2], g2, nullptr, 0, 0, 0);
  gemm(g2, 528, 256, dw[3], 64, dbp[3], nullptr, nullptr, out_dec, 57, 57, 1);
}
